// MultiHeadAttention_86629490360910
// MI455X (gfx1250) — matrix operand order repaired and checked by reading, not compiled, not run on hardware
//
#include <hip/hip_runtime.h>
#include <hip/hip_bf16.h>

// Problem constants (from reference)
#define B_   2
#define S_   2048
#define D_   768
#define H_   12
#define DK_  64
#define LOG512 6.2383246250395077f   // ln(512)

typedef _Float16 half_t;
typedef __attribute__((ext_vector_type(16))) _Float16 v16h;
typedef __attribute__((ext_vector_type(8)))  _Float16 v8h;
typedef __attribute__((ext_vector_type(8)))  float    v8f;

// ---------------------------------------------------------------------------
// Fragment helpers (layouts per CDNA5 ISA §7.12.2, wave32)
// ---------------------------------------------------------------------------

// Load 16 halves from two contiguous 16B chunks.
__device__ __forceinline__ v16h ld2x8(const half_t* p0, const half_t* p1) {
  v8h a = *(const v8h*)p0;
  v8h b = *(const v8h*)p1;
  v16h r;
#pragma unroll
  for (int i = 0; i < 8; ++i) { r[i] = a[i]; r[i + 8] = b[i]; }
  return r;
}

// A-operand fragment: 16x32 f16 tile from row-major src (leading dim ld),
// top-left (m0,k0). Lane L: M = L%16; K-base = (L<16)?0:8;
// elements 0..7 -> K = kb..kb+7, elements 8..15 -> K = kb+16..kb+23.
__device__ __forceinline__ v16h ldA(const half_t* a, int ld, int m0, int k0, int lane) {
  int m  = m0 + (lane & 15);
  int kb = k0 + ((lane & 16) ? 8 : 0);
  const half_t* p = a + (size_t)m * ld + kb;
  return ld2x8(p, p + 16);
}

// B-operand fragment: 32x16 f16 tile; element (k,n) stored K-contiguous at
// bt[n*ld + k]. Lane L: N = L%16; K-base = (L<16)?0:8; elements 0..7 -> kb..kb+7, 8..15 -> kb+16..kb+23.
__device__ __forceinline__ v16h ldB(const half_t* bt, int ld, int n0, int k0, int lane) {
  int n  = n0 + (lane & 15);
  int kb = k0 + ((lane & 16) ? 8 : 0);
  const half_t* p = bt + (size_t)n * ld + kb;
  return ld2x8(p, p + 16);
}

__device__ __forceinline__ v8f wmma_f16(v16h a, v16h b, v8f c) {
  return __builtin_amdgcn_wmma_f32_16x16x32_f16(false, a, false, b, (short)0, c,
                                                false, false);
}

// ---------------------------------------------------------------------------
// Conversion kernels
// ---------------------------------------------------------------------------
__global__ void cvt_f32_f16(const float* __restrict__ src, half_t* __restrict__ dst, int n) {
  int i = blockIdx.x * blockDim.x + threadIdx.x;
  if (i < n) dst[i] = (half_t)src[i];
}

// (rows x cols) fp32 row-major -> f16 transposed (cols x rows)
__global__ void transpose_f16(const float* __restrict__ src, half_t* __restrict__ dst,
                              int rows, int cols) {
  int i = blockIdx.x * blockDim.x + threadIdx.x;
  if (i < rows * cols) {
    int r = i / cols, c = i - r * cols;
    dst[(size_t)c * rows + r] = (half_t)src[i];
  }
}

// Per-batch softmax-plus temperature: lam[b] = log(max(sum(v_mask[b]),1))/log(512)
__global__ void lambda_kernel(const int* __restrict__ v_mask, float* __restrict__ lam) {
  __shared__ int sbuf[256];
  int b = blockIdx.x, t = threadIdx.x;
  int sum = 0;
  for (int i = t; i < S_; i += 256) sum += v_mask[b * S_ + i];
  sbuf[t] = sum;
  __syncthreads();
  for (int o = 128; o > 0; o >>= 1) {
    if (t < o) sbuf[t] += sbuf[t + o];
    __syncthreads();
  }
  if (t == 0) {
    int l = sbuf[0] > 1 ? sbuf[0] : 1;
    lam[b] = __logf((float)l) / LOG512;
  }
}

// ---------------------------------------------------------------------------
// WMMA GEMM:  out = A(MxK,f16 row-major) @ B(KxN, stored transposed f16) + bias
// Each wave computes a 32x32 output block (2x2 register-blocked WMMA tiles),
// so each fragment load feeds two WMMAs (1.0 fragment loads per WMMA).
// ---------------------------------------------------------------------------
#define GM_MODE_F16 0   // write f16 row-major (MxN)
#define GM_MODE_VT  1   // write f16 as (B,H,64,S) transposed (for V path)
#define GM_MODE_F32 2   // write f32 row-major (final output)

__device__ __forceinline__ void store_tile(v8f c, int m0, int n0, int lane,
                                           const float* __restrict__ bias,
                                           void* __restrict__ out, int N, int mode) {
  int col   = n0 + (lane & 15);
  float bb  = bias[col];
  int rbase = m0 + ((lane & 16) ? 8 : 0);
#pragma unroll
  for (int r = 0; r < 8; ++r) {
    int row = rbase + r;
    float val = c[r] + bb;
    if (mode == GM_MODE_F16) {
      ((half_t*)out)[(size_t)row * N + col] = (half_t)val;
    } else if (mode == GM_MODE_VT) {
      int bi = row / S_;  int s = row - bi * S_;
      int h  = col / DK_; int d = col - h * DK_;
      ((half_t*)out)[(((size_t)bi * H_ + h) * DK_ + d) * S_ + s] = (half_t)val;
    } else {
      ((float*)out)[(size_t)row * N + col] = val;
    }
  }
}

__global__ __launch_bounds__(256)
void gemm_wmma(const half_t* __restrict__ A, const half_t* __restrict__ BT,
               const float* __restrict__ bias, void* __restrict__ out,
               int M, int N, int K, int mode) {
  const int lane = threadIdx.x & 31;
  const int wv   = threadIdx.x >> 5;
  const int tiles_n = N >> 5;               // 32-wide output blocks
  int flat = blockIdx.x * 8 + wv;
  int m0 = (flat / tiles_n) << 5;
  int n0 = (flat % tiles_n) << 5;

  // Per-lane row base pointers (K-contiguous in both A and BT).
  const half_t* arow0 = A  + (size_t)(m0 + (lane & 15)) * K;
  const half_t* arow1 = arow0 + (size_t)16 * K;
  const half_t* brow0 = BT + (size_t)(n0 + (lane & 15)) * K;
  const half_t* brow1 = brow0 + (size_t)16 * K;
  const int ka = (lane & 16) ? 8 : 0;       // A-fragment K sub-offset
  const int kb = (lane & 16) ? 8 : 0;       // B-fragment K sub-offset

  v8f c00 = {}, c01 = {}, c10 = {}, c11 = {};
  for (int k0 = 0; k0 < K; k0 += 32) {
    v16h a0 = ld2x8(arow0 + k0 + ka, arow0 + k0 + ka + 16);
    v16h a1 = ld2x8(arow1 + k0 + ka, arow1 + k0 + ka + 16);
    v16h b0 = ld2x8(brow0 + k0 + kb, brow0 + k0 + kb + 16);
    v16h b1 = ld2x8(brow1 + k0 + kb, brow1 + k0 + kb + 16);
    c00 = wmma_f16(a0, b0, c00);
    c01 = wmma_f16(a0, b1, c01);
    c10 = wmma_f16(a1, b0, c10);
    c11 = wmma_f16(a1, b1, c11);
    // Prefetch two k-steps ahead (global_prefetch_b8; speculative, safe).
    __builtin_prefetch(arow0 + k0 + 64, 0, 1);
    __builtin_prefetch(arow1 + k0 + 64, 0, 1);
    __builtin_prefetch(brow0 + k0 + 64, 0, 1);
    __builtin_prefetch(brow1 + k0 + 64, 0, 1);
  }

  store_tile(c00, m0,      n0,      lane, bias, out, N, mode);
  store_tile(c01, m0,      n0 + 16, lane, bias, out, N, mode);
  store_tile(c10, m0 + 16, n0,      lane, bias, out, N, mode);
  store_tile(c11, m0 + 16, n0 + 16, lane, bias, out, N, mode);
}

// ---------------------------------------------------------------------------
// Flash attention with masked softmax-plus.
// Block = 256 threads = 8 waves; each wave owns a 16-query tile of one (b,h).
// qw,kw: (B,S,H*64) f16 row-major.  vt: (B,H,64,S) f16.  attn: (B,S,H*64) f16.
// ---------------------------------------------------------------------------
__global__ __launch_bounds__(256)
void flash_attn(const half_t* __restrict__ qw, const half_t* __restrict__ kw,
                const half_t* __restrict__ vt, const float* __restrict__ lam,
                const int* __restrict__ v_mask, half_t* __restrict__ attn) {
  __shared__ __align__(16) half_t plds[8][16 * 32];

  const int lane = threadIdx.x & 31;
  const int wv   = threadIdx.x >> 5;
  const int jtiles = S_ / 128;          // query tiles of 128 rows per block
  int bh = blockIdx.x / jtiles;
  int jt = blockIdx.x % jtiles;
  int b = bh / H_, h = bh % H_;
  int j0 = jt * 128 + wv * 16;

  const half_t* qb  = qw + (size_t)b * S_ * D_ + h * DK_;
  const half_t* kwp = kw + (size_t)b * S_ * D_ + h * DK_;
  const half_t* vb  = vt + ((size_t)b * H_ + h) * (size_t)DK_ * S_;
  const int*    msk = v_mask + b * S_;
  float scale = lam[b] * 0.125f;        // lambda / sqrt(64)

  // Q fragments for the whole head dim (d 0..31 and 32..63), kept in VGPRs.
  v16h qa0 = ldA(qb, D_, j0, 0,  lane);
  v16h qa1 = ldA(qb, D_, j0, 32, lane);

  float mrow[8], lrow[8];
  v8f o0 = {}, o1 = {}, o2 = {}, o3 = {};
#pragma unroll
  for (int r = 0; r < 8; ++r) { mrow[r] = -1e30f; lrow[r] = 0.f; }

  half_t* pl = &plds[wv][0];

  for (int k0 = 0; k0 < S_; k0 += 32) {
    // S = Q K^T for 32 keys: two 16x16 f32 tiles, each summed over d in 2 WMMAs.
    v8f c0 = {}, c1 = {};
    c0 = wmma_f16(qa0, ldB(kwp, D_, k0,      0,  lane), c0);
    c0 = wmma_f16(qa1, ldB(kwp, D_, k0,      32, lane), c0);
    c1 = wmma_f16(qa0, ldB(kwp, D_, k0 + 16, 0,  lane), c1);
    c1 = wmma_f16(qa1, ldB(kwp, D_, k0 + 16, 32, lane), c1);

    // Masked softmax-plus logits: valid -> s*lambda/8, masked -> -1e30.
    int key0 = k0 + (lane & 15), key1 = key0 + 16;
    float mul0 = msk[key0] ? scale : 0.f;
    float add0 = msk[key0] ? 0.f   : -1e30f;
    float mul1 = msk[key1] ? scale : 0.f;
    float add1 = msk[key1] ? 0.f   : -1e30f;

    float p0[8], p1[8];
#pragma unroll
    for (int r = 0; r < 8; ++r) {
      float s0 = c0[r] * mul0 + add0;
      float s1 = c1[r] * mul1 + add1;
      // chunk row-max across the 16-lane group holding row M = r + (lane>=16)*8
      float cm = fmaxf(s0, s1);
      cm = fmaxf(cm, __shfl_xor(cm, 1, 32));
      cm = fmaxf(cm, __shfl_xor(cm, 2, 32));
      cm = fmaxf(cm, __shfl_xor(cm, 4, 32));
      cm = fmaxf(cm, __shfl_xor(cm, 8, 32));
      float mn   = fmaxf(mrow[r], cm);
      float corr = __expf(mrow[r] - mn);
      mrow[r] = mn;
      p0[r] = __expf(s0 - mn);
      p1[r] = __expf(s1 - mn);
      float ps = p0[r] + p1[r];
      ps += __shfl_xor(ps, 1, 32);
      ps += __shfl_xor(ps, 2, 32);
      ps += __shfl_xor(ps, 4, 32);
      ps += __shfl_xor(ps, 8, 32);
      lrow[r] = lrow[r] * corr + ps;
      o0[r] *= corr; o1[r] *= corr; o2[r] *= corr; o3[r] *= corr;
    }

    // Stage P (16x32 f16, row-major) through this wave's private LDS region
    // to re-layout from C-fragment to A-fragment.
    {
      int prow = (lane & 16) ? 8 : 0;
      int pcol = lane & 15;
#pragma unroll
      for (int r = 0; r < 8; ++r) {
        pl[(prow + r) * 32 + pcol]      = (half_t)p0[r];
        pl[(prow + r) * 32 + 16 + pcol] = (half_t)p1[r];
      }
    }
    asm volatile("s_wait_dscnt 0" ::: "memory");   // DS stores visible to reads
    v16h pa;
    {
      int m   = lane & 15;
      int kbs = (lane & 16) ? 8 : 0;
      const half_t* pp = pl + m * 32 + kbs;
      v8h x0 = *(const v8h*)pp;
      v8h x1 = *(const v8h*)(pp + 16);
#pragma unroll
      for (int i = 0; i < 8; ++i) { pa[i] = x0[i]; pa[i + 8] = x1[i]; }
    }

    // O += P @ V  (V is K-contiguous per head-dim row in vt)
    o0 = wmma_f16(pa, ldB(vb, S_, 0,  k0, lane), o0);
    o1 = wmma_f16(pa, ldB(vb, S_, 16, k0, lane), o1);
    o2 = wmma_f16(pa, ldB(vb, S_, 32, k0, lane), o2);
    o3 = wmma_f16(pa, ldB(vb, S_, 48, k0, lane), o3);
  }

  // Normalize by row sums and write f16 row-major into attn (B,S,H*64).
  int col = lane & 15;
  int rb  = (lane & 16) ? 8 : 0;
#pragma unroll
  for (int r = 0; r < 8; ++r) {
    float inv = 1.0f / lrow[r];
    int j = j0 + rb + r;
    size_t base = ((size_t)b * S_ + j) * D_ + h * DK_;
    attn[base + 0 * 16 + col] = (half_t)(o0[r] * inv);
    attn[base + 1 * 16 + col] = (half_t)(o1[r] * inv);
    attn[base + 2 * 16 + col] = (half_t)(o2[r] * inv);
    attn[base + 3 * 16 + col] = (half_t)(o3[r] * inv);
  }
}

// ---------------------------------------------------------------------------
// Host-side launcher
// ---------------------------------------------------------------------------
extern "C" void kernel_launch(void* const* d_in, const int* in_sizes, int n_in,
                              void* d_out, int out_size, void* d_ws, size_t ws_size,
                              hipStream_t stream) {
  (void)in_sizes; (void)n_in; (void)out_size; (void)ws_size;

  const float* q  = (const float*)d_in[0];
  const float* k  = (const float*)d_in[1];
  const float* v  = (const float*)d_in[2];
  const float* Wq = (const float*)d_in[3];
  const float* bq = (const float*)d_in[4];
  const float* Wk = (const float*)d_in[5];
  const float* bk = (const float*)d_in[6];
  const float* Wv = (const float*)d_in[7];
  const float* bv = (const float*)d_in[8];
  const float* Wo = (const float*)d_in[9];
  const float* bo = (const float*)d_in[10];
  const int*   vm = (const int*)d_in[11];

  // Workspace layout (f16 staging buffers), 256B-aligned chunks.
  char* ws = (char*)d_ws;
  size_t off = 0;
  auto alloc = [&](size_t bytes) -> char* {
    char* p = ws + off;
    off += bytes;
    off = (off + 255) & ~(size_t)255;
    return p;
  };
  const size_t nX = (size_t)B_ * S_ * D_;   // 3,145,728 elements
  const size_t nW = (size_t)D_ * D_;        //   589,824 elements

  half_t* q16    = (half_t*)alloc(nX * 2);
  half_t* k16    = (half_t*)alloc(nX * 2);
  half_t* v16    = (half_t*)alloc(nX * 2);
  half_t* wqT    = (half_t*)alloc(nW * 2);
  half_t* wkT    = (half_t*)alloc(nW * 2);
  half_t* wvT    = (half_t*)alloc(nW * 2);
  half_t* woT    = (half_t*)alloc(nW * 2);
  half_t* qw16   = (half_t*)alloc(nX * 2);
  half_t* kw16   = (half_t*)alloc(nX * 2);
  half_t* vwT16  = (half_t*)alloc(nX * 2);   // (B,H,64,S)
  half_t* attn16 = (half_t*)alloc(nX * 2);
  float*  lam    = (float*)alloc(256);

  // 1) fp32 -> f16 conversions
  int cvt_blk = (int)((nX + 255) / 256);
  cvt_f32_f16<<<cvt_blk, 256, 0, stream>>>(q, q16, (int)nX);
  cvt_f32_f16<<<cvt_blk, 256, 0, stream>>>(k, k16, (int)nX);
  cvt_f32_f16<<<cvt_blk, 256, 0, stream>>>(v, v16, (int)nX);
  int tr_blk = (int)((nW + 255) / 256);
  transpose_f16<<<tr_blk, 256, 0, stream>>>(Wq, wqT, D_, D_);
  transpose_f16<<<tr_blk, 256, 0, stream>>>(Wk, wkT, D_, D_);
  transpose_f16<<<tr_blk, 256, 0, stream>>>(Wv, wvT, D_, D_);
  transpose_f16<<<tr_blk, 256, 0, stream>>>(Wo, woT, D_, D_);

  // 2) per-batch softmax-plus temperature
  lambda_kernel<<<B_, 256, 0, stream>>>(vm, lam);

  // 3) QKV projections via WMMA GEMM (M=4096, N=768, K=768), 32x32 per wave
  const int M = B_ * S_;
  const int gemm_blocks = ((M / 32) * (D_ / 32)) / 8;   // 384 blocks, 8 waves each
  gemm_wmma<<<gemm_blocks, 256, 0, stream>>>(q16, wqT, bq, qw16,  M, D_, D_, GM_MODE_F16);
  gemm_wmma<<<gemm_blocks, 256, 0, stream>>>(k16, wkT, bk, kw16,  M, D_, D_, GM_MODE_F16);
  gemm_wmma<<<gemm_blocks, 256, 0, stream>>>(v16, wvT, bv, vwT16, M, D_, D_, GM_MODE_VT);

  // 4) flash attention (one wave per 16-query tile; 8 waves/block)
  flash_attn<<<B_ * H_ * (S_ / 128), 256, 0, stream>>>(qw16, kw16, vwT16, lam, vm, attn16);

  // 5) output projection, fp32 result straight into d_out
  gemm_wmma<<<gemm_blocks, 256, 0, stream>>>(attn16, woT, bo, d_out, M, D_, D_, GM_MODE_F32);
}
